// FastShapeTransformerModel_40647570489995
// MI455X (gfx1250) — matrix operand order repaired and checked by reading, not compiled, not run on hardware
//
#include <hip/hip_runtime.h>
#include <hip/hip_bf16.h>
#include <math.h>

// ---------------------------------------------------------------------------
// Model constants (from the reference)
// ---------------------------------------------------------------------------
#define LAYERS 4
#define EMB    512
#define HEADS  8
#define HDIM   64
#define NB     2
#define SEQ    1024
#define FFD    2048
#define VOC1   17          // V+1
#define SPV    65          // 2^TD+1
#define ADIM   3
#define TOK    (NB*SEQ)    // 2048 tokens

typedef __attribute__((ext_vector_type(16))) __bf16 v16bf;
typedef __attribute__((ext_vector_type(8)))  __bf16 v8bf;
typedef __attribute__((ext_vector_type(4)))  __bf16 v4bf;
typedef __attribute__((ext_vector_type(8)))  float  v8f;

// ---------------------------------------------------------------------------
// fp32 -> bf16 bulk conversion (weights; run once per call, stays in L2)
// ---------------------------------------------------------------------------
__global__ __launch_bounds__(256) void cvt_bf16_kernel(
    const float* __restrict__ in, __bf16* __restrict__ out, int n)
{
    int idx = (blockIdx.x * 256 + threadIdx.x) * 4;
    if (idx + 3 < n) {
        float4 f = *(const float4*)(in + idx);
        v4bf o;
        o[0] = (__bf16)f.x; o[1] = (__bf16)f.y;
        o[2] = (__bf16)f.z; o[3] = (__bf16)f.w;
        *(v4bf*)(out + idx) = o;
    } else {
        for (int i = idx; i < n; ++i) out[i] = (__bf16)in[i];
    }
}

// ---------------------------------------------------------------------------
// Fused embedding sum + right-shift with SOS token; emits fp32 + bf16 copies
// ---------------------------------------------------------------------------
__global__ __launch_bounds__(256) void embed_kernel(
    const int* __restrict__ seq, const int* __restrict__ depth,
    const int* __restrict__ pos, const float* __restrict__ tok,
    const float* __restrict__ dep, const float* __restrict__ sp,
    const float* __restrict__ sos, float* __restrict__ xf,
    __bf16* __restrict__ xb)
{
    int idx = blockIdx.x * 256 + threadIdx.x;
    if (idx >= TOK * EMB) return;
    int e  = idx & (EMB - 1);
    int ns = idx >> 9;           // log2(EMB)=9
    int s  = ns & (SEQ - 1);
    int n  = ns >> 10;           // log2(SEQ)=10
    float val;
    if (s == 0) {
        val = sos[e];
    } else {
        int t  = s - 1;
        int tv = seq[n * SEQ + t];
        int dv = depth[n * SEQ + t];
        val = tok[tv * EMB + e] + dep[dv * EMB + e];
#pragma unroll
        for (int a = 0; a < ADIM; ++a) {
            int p = pos[(a * NB + n) * SEQ + t];
            val += sp[((size_t)a * SPV + p) * EMB + e];
        }
    }
    xf[idx] = val;
    xb[idx] = (__bf16)val;
}

// ---------------------------------------------------------------------------
// bf16-resident WMMA GEMM: out = act(X[M][K] @ W[Nout][K]^T + bias) (+res)
//
// X and W are bf16, K-major -> every lane's WMMA fragment is two contiguous
// 16-byte chunks: 2x global_load_b128 per fragment, assembled into v16bf by
// __builtin_shufflevector (register coalescing, no ALU). No LDS, no barriers,
// no conversions in the hot loop: 8 b128 loads + 4 v_wmma per K-step.
//  - A 16x32 layout: lane holds K {khi..khi+7, khi+16..khi+23} of row m
//  - B 32x16 layout: lane holds K {kb..kb+7, kb+16..kb+23} of column n (= W row n)
// Each wave computes a 32x32 macro-tile (2x2 WMMA tiles); block = 8 waves
// (2M x 4N) -> 64x128 tile. OOB rows/cols CLAMPED for loads (stores guarded)
// so the hot loop is branch-free with EXEC all-ones (WMMA requirement).
// act: 0 = none, 1 = exact GELU.  outf / outb each optional.
// ---------------------------------------------------------------------------
__global__ __launch_bounds__(256) void gemm_bf16_wmma(
    const __bf16* __restrict__ X, const __bf16* __restrict__ W,
    const float* __restrict__ bias, const float* __restrict__ res,
    float* __restrict__ outf, __bf16* __restrict__ outb,
    int M, int Nout, int K, int act)
{
    const int tid  = threadIdx.x;
    const int lane = tid & 31;
    const int wave = tid >> 5;                              // 0..7
    const int blockM = blockIdx.y * 64 + (wave >> 2) * 32;  // wave M origin
    const int blockN = blockIdx.x * 128 + (wave & 3) * 32;  // wave N origin

    const int fm   = lane & 15;                 // row (A) / col (B) within 16-tile
    const int hi   = lane >> 4;                 // lane half
    const int khiA = hi * 8;                    // A: lanes 16-31 start at K+8 / K+24
    const int kbB  = hi * 8;                    // B: lanes 16-31 start at K+8 / K+24

    // clamped row pointers (clamp-garbage never stored)
    int rA0 = blockM + fm;       if (rA0 >= M)    rA0 = M - 1;
    int rA1 = blockM + 16 + fm;  if (rA1 >= M)    rA1 = M - 1;
    int rB0 = blockN + fm;       if (rB0 >= Nout) rB0 = Nout - 1;
    int rB1 = blockN + 16 + fm;  if (rB1 >= Nout) rB1 = Nout - 1;

    const v8bf* A0 = (const v8bf*)(X + (size_t)rA0 * K);
    const v8bf* A1 = (const v8bf*)(X + (size_t)rA1 * K);
    const v8bf* B0 = (const v8bf*)(W + (size_t)rB0 * K);
    const v8bf* B1 = (const v8bf*)(W + (size_t)rB1 * K);

    v8f c00 = {0.f,0.f,0.f,0.f,0.f,0.f,0.f,0.f};
    v8f c01 = {0.f,0.f,0.f,0.f,0.f,0.f,0.f,0.f};
    v8f c10 = {0.f,0.f,0.f,0.f,0.f,0.f,0.f,0.f};
    v8f c11 = {0.f,0.f,0.f,0.f,0.f,0.f,0.f,0.f};

    for (int kt = 0; kt < K; kt += 32) {
        const int f0 = (kt + khiA) >> 3;        // v8bf index: A k-block 0
        const int fb = (kt + kbB)  >> 3;        // v8bf index: B k-run

        v8bf a0l = A0[f0], a0h = A0[f0 + 2];    // +16 bf16 = +2 v8bf
        v8bf a1l = A1[f0], a1h = A1[f0 + 2];
        v8bf b0l = B0[fb], b0h = B0[fb + 2];
        v8bf b1l = B1[fb], b1h = B1[fb + 2];

        v16bf a0 = __builtin_shufflevector(a0l, a0h, 0,1,2,3,4,5,6,7,8,9,10,11,12,13,14,15);
        v16bf a1 = __builtin_shufflevector(a1l, a1h, 0,1,2,3,4,5,6,7,8,9,10,11,12,13,14,15);
        v16bf b0 = __builtin_shufflevector(b0l, b0h, 0,1,2,3,4,5,6,7,8,9,10,11,12,13,14,15);
        v16bf b1 = __builtin_shufflevector(b1l, b1h, 0,1,2,3,4,5,6,7,8,9,10,11,12,13,14,15);

        c00 = __builtin_amdgcn_wmma_f32_16x16x32_bf16(false, a0, false, b0, (short)0, c00, false, false);
        c01 = __builtin_amdgcn_wmma_f32_16x16x32_bf16(false, a0, false, b1, (short)0, c01, false, false);
        c10 = __builtin_amdgcn_wmma_f32_16x16x32_bf16(false, a1, false, b0, (short)0, c10, false, false);
        c11 = __builtin_amdgcn_wmma_f32_16x16x32_bf16(false, a1, false, b1, (short)0, c11, false, false);
    }

    // epilogue: C/D layout -> lane col = lane&15, rows r + 8*hi within 16-tile
    auto store_tile = [&](const v8f& c, int mo, int no) {
        int col = blockN + no + fm;
        if (col >= Nout) return;
        float bval = bias ? bias[col] : 0.f;
#pragma unroll
        for (int r = 0; r < 8; ++r) {
            int gm = blockM + mo + hi * 8 + r;
            if (gm < M) {
                float val = c[r] + bval;
                if (act == 1)  // exact GELU
                    val = 0.5f * val * (1.0f + erff(val * 0.7071067811865476f));
                if (res) val += res[(size_t)gm * Nout + col];
                size_t o = (size_t)gm * Nout + col;
                if (outf) outf[o] = val;
                if (outb) outb[o] = (__bf16)val;
            }
        }
    };
    store_tile(c00,  0,  0);
    store_tile(c01,  0, 16);
    store_tile(c10, 16,  0);
    store_tile(c11, 16, 16);
}

// ---------------------------------------------------------------------------
// Causal linear attention recurrence, one block per (n,h). Emits bf16
// (consumed only by the Wo WMMA GEMM).
// ---------------------------------------------------------------------------
__global__ __launch_bounds__(256) void linattn_kernel(
    const float* __restrict__ q, const float* __restrict__ k,
    const float* __restrict__ v, const int* __restrict__ seq,
    __bf16* __restrict__ att)
{
    __shared__ float kv[HDIM][HDIM + 1];
    __shared__ float ksum[HDIM];
    __shared__ float qf[HDIM], kf[HDIM], vv[HDIM];
    __shared__ float part[4][HDIM];
    __shared__ float partZ[HDIM];

    const int b = blockIdx.x;
    const int n = b / HEADS;
    const int h = b % HEADS;
    const int t = threadIdx.x;

    for (int e = t; e < HDIM * (HDIM + 1); e += 256) (&kv[0][0])[e] = 0.f;
    if (t < HDIM) ksum[t] = 0.f;
    __syncthreads();

    for (int s = 0; s < SEQ; ++s) {
        if (t < HDIM) {
            size_t base = ((size_t)n * SEQ + s) * EMB + h * HDIM + t;
            float qd = q[base], kd = k[base];
            float msk = (seq[n * SEQ + s] != 0) ? 1.f : 0.f;
            qf[t] = (qd > 0.f) ? (qd + 1.f) : __expf(qd);   // elu(x)+1
            kf[t] = ((kd > 0.f) ? (kd + 1.f) : __expf(kd)) * msk;
            vv[t] = v[base];
        }
        __syncthreads();
        {   // state update: thread t owns d = t>>2, m in [(t&3)*16, +16)
            int d = t >> 2, m0 = (t & 3) * 16;
            float kfd = kf[d];
#pragma unroll
            for (int j = 0; j < 16; ++j) kv[d][m0 + j] += kfd * vv[m0 + j];
            if ((t & 3) == 0) ksum[d] += kfd;
        }
        __syncthreads();
        {   // att partials: thread t owns m = t>>2, d in [(t&3)*16, +16)
            int m = t >> 2, d0 = (t & 3) * 16;
            float acc = 0.f;
#pragma unroll
            for (int j = 0; j < 16; ++j) acc += qf[d0 + j] * kv[d0 + j][m];
            part[t & 3][m] = acc;
        }
        if (t < HDIM) partZ[t] = qf[t] * ksum[t];
        __syncthreads();
        if (t < HDIM) {
            float a = part[0][t] + part[1][t] + part[2][t] + part[3][t];
            float z = 0.f;
#pragma unroll 8
            for (int j = 0; j < HDIM; ++j) z += partZ[j];   // broadcast LDS reads
            size_t base = ((size_t)n * SEQ + s) * EMB + h * HDIM + t;
            att[base] = (__bf16)(a / (z + 1e-6f));
        }
        __syncthreads();
    }
}

// ---------------------------------------------------------------------------
// LayerNorm over E=512, one 128-thread block per token; fp32 + bf16 outputs.
// ---------------------------------------------------------------------------
__global__ __launch_bounds__(128) void layernorm_kernel(
    const float* __restrict__ x, const float* __restrict__ g,
    const float* __restrict__ bb, float* __restrict__ outf,
    __bf16* __restrict__ outb)
{
    __shared__ float red[128];
    const int row = blockIdx.x;
    const int t   = threadIdx.x;
    const float* xr = x + (size_t)row * EMB;

    float v0 = xr[t], v1 = xr[t + 128], v2 = xr[t + 256], v3 = xr[t + 384];
    red[t] = v0 + v1 + v2 + v3;
    __syncthreads();
    for (int off = 64; off > 0; off >>= 1) {
        if (t < off) red[t] += red[t + off];
        __syncthreads();
    }
    float mean = red[0] * (1.f / EMB);
    __syncthreads();
    float d0 = v0 - mean, d1 = v1 - mean, d2 = v2 - mean, d3 = v3 - mean;
    red[t] = d0 * d0 + d1 * d1 + d2 * d2 + d3 * d3;
    __syncthreads();
    for (int off = 64; off > 0; off >>= 1) {
        if (t < off) red[t] += red[t + off];
        __syncthreads();
    }
    float rstd = rsqrtf(red[0] * (1.f / EMB) + 1e-5f);
    float y0 = d0 * rstd * g[t]       + bb[t];
    float y1 = d1 * rstd * g[t + 128] + bb[t + 128];
    float y2 = d2 * rstd * g[t + 256] + bb[t + 256];
    float y3 = d3 * rstd * g[t + 384] + bb[t + 384];
    float*  of = outf + (size_t)row * EMB;
    __bf16* ob = outb + (size_t)row * EMB;
    of[t] = y0; of[t + 128] = y1; of[t + 256] = y2; of[t + 384] = y3;
    ob[t] = (__bf16)y0; ob[t + 128] = (__bf16)y1;
    ob[t + 256] = (__bf16)y2; ob[t + 384] = (__bf16)y3;
}

// ---------------------------------------------------------------------------
// Host-side pipeline
// ---------------------------------------------------------------------------
extern "C" void kernel_launch(void* const* d_in, const int* in_sizes, int n_in,
                              void* d_out, int out_size, void* d_ws, size_t ws_size,
                              hipStream_t stream)
{
    (void)in_sizes; (void)n_in; (void)out_size; (void)ws_size;

    const int*   seq   = (const int*)  d_in[0];
    const int*   depth = (const int*)  d_in[1];
    const int*   pos   = (const int*)  d_in[2];
    const float* tok   = (const float*)d_in[3];
    const float* dep   = (const float*)d_in[4];
    const float* sp    = (const float*)d_in[5];
    const float* sos   = (const float*)d_in[6];
    const float* Wq    = (const float*)d_in[7];
    const float* bq    = (const float*)d_in[8];
    const float* Wk    = (const float*)d_in[9];
    const float* bk    = (const float*)d_in[10];
    const float* Wv    = (const float*)d_in[11];
    const float* bv    = (const float*)d_in[12];
    const float* Wo    = (const float*)d_in[13];
    const float* bo    = (const float*)d_in[14];
    const float* g1    = (const float*)d_in[15];
    const float* b1    = (const float*)d_in[16];
    const float* W1    = (const float*)d_in[17];
    const float* bf1   = (const float*)d_in[18];
    const float* W2    = (const float*)d_in[19];
    const float* bf2   = (const float*)d_in[20];
    const float* g2    = (const float*)d_in[21];
    const float* b2    = (const float*)d_in[22];
    const float* gf    = (const float*)d_in[23];
    const float* bfin  = (const float*)d_in[24];
    const float* Whead = (const float*)d_in[25];

    // ---- workspace carve-up (256B aligned) ----
    char* base = (char*)d_ws;
    auto alloc = [&](size_t bytes) -> char* {
        char* p = base;
        base += (bytes + 255) & ~(size_t)255;
        return p;
    };
    const size_t TE = (size_t)TOK * EMB;
    float*  xf  = (float*) alloc(TE * 4);
    __bf16* xb  = (__bf16*)alloc(TE * 2);
    float*  qf  = (float*) alloc(TE * 4);
    float*  kf  = (float*) alloc(TE * 4);
    float*  vf  = (float*) alloc(TE * 4);
    __bf16* ab  = (__bf16*)alloc(TE * 2);
    float*  tbf = (float*) alloc(TE * 4);
    __bf16* tbb = (__bf16*)alloc(TE * 2);
    __bf16* ffb = (__bf16*)alloc((size_t)TOK * FFD * 2);
    const size_t WSQ = (size_t)LAYERS * EMB * EMB;   // 1,048,576
    const size_t WSF = (size_t)LAYERS * FFD * EMB;   // 4,194,304
    __bf16* WqB = (__bf16*)alloc(WSQ * 2);
    __bf16* WkB = (__bf16*)alloc(WSQ * 2);
    __bf16* WvB = (__bf16*)alloc(WSQ * 2);
    __bf16* WoB = (__bf16*)alloc(WSQ * 2);
    __bf16* W1B = (__bf16*)alloc(WSF * 2);
    __bf16* W2B = (__bf16*)alloc(WSF * 2);
    __bf16* WhB = (__bf16*)alloc((size_t)VOC1 * EMB * 2);

    // ---- weights -> bf16 (once per call; then resident in L2) ----
    auto cvt = [&](const float* in, __bf16* out, size_t n) {
        cvt_bf16_kernel<<<(unsigned)((n / 4 + 255) / 256), 256, 0, stream>>>(in, out, (int)n);
    };
    cvt(Wq, WqB, WSQ);  cvt(Wk, WkB, WSQ);  cvt(Wv, WvB, WSQ);  cvt(Wo, WoB, WSQ);
    cvt(W1, W1B, WSF);  cvt(W2, W2B, WSF);  cvt(Whead, WhB, (size_t)VOC1 * EMB);

    // ---- embeddings + shift ----
    embed_kernel<<<(TOK * EMB + 255) / 256, 256, 0, stream>>>(
        seq, depth, pos, tok, dep, sp, sos, xf, xb);

    auto gemm = [&](const __bf16* Xp, const __bf16* Wp, const float* bp,
                    const float* rp, float* of, __bf16* ob,
                    int M, int Nn, int K, int act) {
        dim3 grid((Nn + 127) / 128, (M + 63) / 64);
        gemm_bf16_wmma<<<grid, dim3(256), 0, stream>>>(Xp, Wp, bp, rp, of, ob, M, Nn, K, act);
    };

    for (int l = 0; l < LAYERS; ++l) {
        const __bf16* Wql = WqB + (size_t)l * EMB * EMB;
        const __bf16* Wkl = WkB + (size_t)l * EMB * EMB;
        const __bf16* Wvl = WvB + (size_t)l * EMB * EMB;
        const __bf16* Wol = WoB + (size_t)l * EMB * EMB;
        const __bf16* W1l = W1B + (size_t)l * FFD * EMB;
        const __bf16* W2l = W2B + (size_t)l * EMB * FFD;

        gemm(xb, Wql, bq + l * EMB, nullptr, qf, nullptr, TOK, EMB, EMB, 0);
        gemm(xb, Wkl, bk + l * EMB, nullptr, kf, nullptr, TOK, EMB, EMB, 0);
        gemm(xb, Wvl, bv + l * EMB, nullptr, vf, nullptr, TOK, EMB, EMB, 0);

        linattn_kernel<<<NB * HEADS, 256, 0, stream>>>(qf, kf, vf, seq, ab);

        gemm(ab, Wol, bo + l * EMB, xf, tbf, nullptr, TOK, EMB, EMB, 0);   // + residual x
        layernorm_kernel<<<TOK, 128, 0, stream>>>(tbf, g1 + l * EMB, b1 + l * EMB, xf, xb);

        gemm(xb, W1l, bf1 + l * FFD, nullptr, nullptr, ffb, TOK, FFD, EMB, 1);  // GELU, bf16 out
        gemm(ffb, W2l, bf2 + l * EMB, xf, tbf, nullptr, TOK, EMB, FFD, 0);      // + residual x
        layernorm_kernel<<<TOK, 128, 0, stream>>>(tbf, g2 + l * EMB, b2 + l * EMB, xf, xb);
    }

    // final LN + head projection -> logits [N,S,17] (fp32)
    layernorm_kernel<<<TOK, 128, 0, stream>>>(xf, gf, bfin, tbf, tbb);
    gemm(tbb, WhB, nullptr, nullptr, (float*)d_out, nullptr, TOK, VOC1, EMB, 0);
}
